// TFSwiftFormerEfficientAdditiveAttention_32530082300321
// MI455X (gfx1250) — matrix operand order repaired and checked by reading, not compiled, not run on hardware
//
#include <hip/hip_runtime.h>

typedef __attribute__((ext_vector_type(4)))  float   v4f;
typedef __attribute__((ext_vector_type(8)))  float   v8f;
typedef __attribute__((ext_vector_type(8)))  __bf16  v8bf;
typedef __attribute__((ext_vector_type(16))) __bf16  v16bf;

#define MODE_QNORM 0   // Q = xb@wqT + bq ; l2norm ; write f32 qn ; accumulate gq
#define MODE_KNORM 1   // K = xb@wkT + bk ; l2norm ; write bf16 kn
#define MODE_GEMM2 2   // tmp = (gq*kn)@wpT + bp + qn ; write bf16 tmp
#define MODE_GEMM3 3   // out = tmp@wfT + bf ; write f32 out

// ---------------------------------------------------------------------------
// Fused 32x512 (full-row) bf16 WMMA GEMM, K=512, with mode-specific epilogues.
// Block = 256 threads = 8 waves partitioning the 512 columns (64 each).
// Each wave owns 2 m-tiles x 4 n-tiles = 8 WMMA 16x16 tiles (f32 accum).
// Per k-step: 2 A-fragments + 4 B-fragments feed 8 WMMAs (6 loads / 8 MMAs),
// and each B fragment is shared by the two m-tiles -> back-to-back WMMA pairs
// per loaded operand. B columns are wave-exclusive, so each block reads each
// weight element exactly once (L2-resident, 512 KB per GEMM per block).
// A fragment (16x32 bf16): lane L holds row L%16, K = 8*(L/16)+{0..7} and +16
//   -> two contiguous 16B loads from row-major bf16 A.
// B fragment (32x16 bf16): lane L holds col L%16, K = 8*(L/16)+{0..7} and +16
//   -> two contiguous 16B loads from the pre-transposed [N][K] bf16 weight.
// ---------------------------------------------------------------------------
template<int MODE>
__global__ __launch_bounds__(256) void gemm512_kernel(
    const __bf16* __restrict__ A,        // [32768][512] bf16, row-major
    const __bf16* __restrict__ Wt,       // [512][512] bf16, N-major (transposed)
    const float*  __restrict__ bias,     // [512] f32
    const float*  __restrict__ gq_in,    // [8][512] f32 (GEMM2 A-scale)
    float*        __restrict__ gq_out,   // [8][512] f32 (QNORM accumulation)
    const float*  __restrict__ addend,   // [32768][512] f32 qn (GEMM2 epilogue add)
    float*        __restrict__ out_f32,  // QNORM / GEMM3 output
    __bf16*       __restrict__ out_bf16) // KNORM / GEMM2 output
{
    __shared__ float ls_tile[16 * 512];  // staging for coalesced writeout (half tile)
    __shared__ float ls_ss[32];          // per-row sum of squares
    __shared__ float ls_col[512];        // per-column sums for gq

    const int tid   = threadIdx.x;
    const int lane  = tid & 31;
    const int wv    = tid >> 5;          // wave id 0..7 -> 64-column slice
    const int l16   = lane & 15;
    const int h     = lane >> 4;

    const int rows0 = blockIdx.x * 32;   // 32 rows per block; 4096 rows/batch
    const int batch = rows0 >> 12;

    const int akb   = h * 8;             // A K-base within 32-chunk
    const int bcol0 = wv * 64;
    const int bkb   = h * 8;             // B K-base within 32-chunk

    v8f acc[2][4] = {};

    const __bf16* aptr0 = A  + (size_t)(rows0 + l16)      * 512 + akb;
    const __bf16* aptr1 = A  + (size_t)(rows0 + 16 + l16) * 512 + akb;
    const __bf16* bbase = Wt + (size_t)(bcol0 + l16) * 512 + bkb;
    const float*  gptr  = (MODE == MODE_GEMM2) ? (gq_in + batch * 512 + akb) : nullptr;

    // prologue: first A fragments (both m-tiles)
    v8bf a0[2], a1[2];
    a0[0] = *(const v8bf*)(aptr0);
    a1[0] = *(const v8bf*)(aptr0 + 16);
    a0[1] = *(const v8bf*)(aptr1);
    a1[1] = *(const v8bf*)(aptr1 + 16);

    for (int kk = 0; kk < 512; kk += 32) {
        // B fragments for this k-step (wave-private columns)
        v16bf bfr[4];
        #pragma unroll
        for (int n = 0; n < 4; ++n)
            bfr[n] = __builtin_shufflevector(
                *(const v8bf*)(bbase + (size_t)n * (16 * 512) + kk),
                *(const v8bf*)(bbase + (size_t)n * (16 * 512) + kk + 16),
                0, 1, 2, 3, 4, 5, 6, 7, 8, 9, 10, 11, 12, 13, 14, 15);

        // assemble A operands for this k-step
        v16bf a[2];
        if (MODE == MODE_GEMM2) {
            v8f g0 = *(const v8f*)(gptr + kk);
            v8f g1 = *(const v8f*)(gptr + kk + 16);
            #pragma unroll
            for (int mt = 0; mt < 2; ++mt) {
                #pragma unroll
                for (int i = 0; i < 8; ++i) {
                    a[mt][i]     = (__bf16)((float)a0[mt][i] * g0[i]);
                    a[mt][i + 8] = (__bf16)((float)a1[mt][i] * g1[i]);
                }
            }
        } else {
            #pragma unroll
            for (int mt = 0; mt < 2; ++mt)
                a[mt] = __builtin_shufflevector(a0[mt], a1[mt],
                            0, 1, 2, 3, 4, 5, 6, 7, 8, 9, 10, 11, 12, 13, 14, 15);
        }

        // software-pipeline: fetch next k-step's A fragments during the chain
        if (kk + 32 < 512) {
            a0[0] = *(const v8bf*)(aptr0 + kk + 32);
            a1[0] = *(const v8bf*)(aptr0 + kk + 48);
            a0[1] = *(const v8bf*)(aptr1 + kk + 32);
            a1[1] = *(const v8bf*)(aptr1 + kk + 48);
        }

        // 8 WMMAs; each loaded B fragment feeds two consecutive WMMAs
        #pragma unroll
        for (int n = 0; n < 4; ++n) {
            acc[0][n] = __builtin_amdgcn_wmma_f32_16x16x32_bf16(
                            false, a[0], false, bfr[n], (short)0, acc[0][n], false, false);
            acc[1][n] = __builtin_amdgcn_wmma_f32_16x16x32_bf16(
                            false, a[1], false, bfr[n], (short)0, acc[1][n], false, false);
        }
    }

    // ---- epilogue: bias -----------------------------------------------------
    #pragma unroll
    for (int n = 0; n < 4; ++n) {
        float b = bias[bcol0 + n * 16 + l16];
        #pragma unroll
        for (int mt = 0; mt < 2; ++mt)
            #pragma unroll
            for (int r = 0; r < 8; ++r) acc[mt][n][r] += b;
    }

    // ---- fused row-wise L2 normalization (Q and K kernels) ------------------
    if (MODE == MODE_QNORM || MODE == MODE_KNORM) {
        if (tid < 32) ls_ss[tid] = 0.0f;
        __syncthreads();
        #pragma unroll
        for (int mt = 0; mt < 2; ++mt) {
            float p[8];
            #pragma unroll
            for (int r = 0; r < 8; ++r) p[r] = 0.0f;
            #pragma unroll
            for (int n = 0; n < 4; ++n)
                #pragma unroll
                for (int r = 0; r < 8; ++r) p[r] += acc[mt][n][r] * acc[mt][n][r];
            #pragma unroll
            for (int r = 0; r < 8; ++r)
                atomicAdd(&ls_ss[mt * 16 + h * 8 + r], p[r]);   // ds_add_f32
        }
        __syncthreads();
        #pragma unroll
        for (int mt = 0; mt < 2; ++mt)
            #pragma unroll
            for (int r = 0; r < 8; ++r) {
                float rs = rsqrtf(fmaxf(ls_ss[mt * 16 + h * 8 + r], 1e-12f));
                #pragma unroll
                for (int n = 0; n < 4; ++n) acc[mt][n][r] *= rs;
            }
    }

    // ---- fused global-query accumulation (softmax over size-1 axis == 1) ----
    if (MODE == MODE_QNORM) {
        ls_col[tid] = 0.0f;
        ls_col[tid + 256] = 0.0f;
        __syncthreads();
        #pragma unroll
        for (int n = 0; n < 4; ++n) {
            float cs = 0.0f;
            #pragma unroll
            for (int mt = 0; mt < 2; ++mt)
                #pragma unroll
                for (int r = 0; r < 8; ++r) cs += acc[mt][n][r];
            atomicAdd(&ls_col[bcol0 + n * 16 + l16], cs);
        }
        __syncthreads();
        atomicAdd(&gq_out[batch * 512 + tid],       ls_col[tid]);
        atomicAdd(&gq_out[batch * 512 + tid + 256], ls_col[tid + 256]);
    }

    // ---- staged, coalesced writeout (two 16-row halves through LDS) ---------
    #pragma unroll
    for (int half = 0; half < 2; ++half) {
        __syncthreads();
        #pragma unroll
        for (int n = 0; n < 4; ++n) {
            int col = bcol0 + n * 16 + l16;
            #pragma unroll
            for (int r = 0; r < 8; ++r)
                ls_tile[(h * 8 + r) * 512 + col] = acc[half][n][r];
        }
        __syncthreads();
        const size_t gbase = (size_t)(rows0 + half * 16) * 512;
        if (MODE == MODE_QNORM || MODE == MODE_GEMM3) {
            float* dst = out_f32 + gbase;
            #pragma unroll
            for (int j = 0; j < 8; ++j) {
                int i4 = tid + 256 * j;                  // 2048 float4 total
                ((v4f*)dst)[i4] = ((const v4f*)ls_tile)[i4];
            }
        } else {
            #pragma unroll
            for (int j = 0; j < 4; ++j) {
                int c = tid + 256 * j;                   // 1024 chunks of 8
                const float* src = ls_tile + c * 8;
                v8bf o;
                #pragma unroll
                for (int i = 0; i < 8; ++i) {
                    float v = src[i];
                    if (MODE == MODE_GEMM2) v += addend[gbase + c * 8 + i];
                    o[i] = (__bf16)v;
                }
                *(v8bf*)(out_bf16 + gbase + c * 8) = o;
            }
        }
    }
}

// ---------------------------------------------------------------------------
// Helpers: f32 -> bf16 conversion, transposed weight conversion, zero-init.
// ---------------------------------------------------------------------------
__global__ __launch_bounds__(256) void cvt_x_kernel(
    const float* __restrict__ in, __bf16* __restrict__ out, int nchunks)
{
    int c = blockIdx.x * 256 + threadIdx.x;   // chunk of 8 elements
    if (c < nchunks) {
        v4f f0 = ((const v4f*)in)[c * 2];
        v4f f1 = ((const v4f*)in)[c * 2 + 1];
        v8bf o;
        #pragma unroll
        for (int i = 0; i < 4; ++i) { o[i] = (__bf16)f0[i]; o[i + 4] = (__bf16)f1[i]; }
        ((v8bf*)out)[c] = o;
    }
}

__global__ __launch_bounds__(256) void cvt_wT_kernel(
    const float* __restrict__ w, __bf16* __restrict__ wt)
{
    int t = blockIdx.x * 256 + threadIdx.x;   // 512*512 outputs, N-major
    wt[t] = (__bf16)w[(size_t)(t & 511) * 512 + (t >> 9)];
}

__global__ __launch_bounds__(256) void zero_kernel(float* __restrict__ p, int n)
{
    int i = blockIdx.x * 256 + threadIdx.x;
    if (i < n) p[i] = 0.0f;
}

// ---------------------------------------------------------------------------
extern "C" void kernel_launch(void* const* d_in, const int* in_sizes, int n_in,
                              void* d_out, int out_size, void* d_ws, size_t ws_size,
                              hipStream_t stream) {
    const float* x   = (const float*)d_in[0];
    const float* wq  = (const float*)d_in[1];
    const float* bq  = (const float*)d_in[2];
    const float* wk  = (const float*)d_in[3];
    const float* bk  = (const float*)d_in[4];
    const float* wp  = (const float*)d_in[5];
    const float* bp  = (const float*)d_in[6];
    const float* wf  = (const float*)d_in[7];
    const float* bfv = (const float*)d_in[8];
    // d_in[9] = w_g is dead: softmax over an axis of size 1 is identically 1.

    const size_t R = 32768, D = 512;

    char* ws = (char*)d_ws;
    size_t off = 0;
    auto alloc = [&](size_t bytes) -> char* {
        char* p = ws + off;
        off += (bytes + 255) & ~(size_t)255;
        return p;
    };
    __bf16* xb  = (__bf16*)alloc(R * D * 2);   // x in bf16
    __bf16* kn  = (__bf16*)alloc(R * D * 2);   // normalized K, bf16
    __bf16* tmp = (__bf16*)alloc(R * D * 2);   // gemm2 output, bf16
    float*  qn  = (float*) alloc(R * D * 4);   // normalized Q, f32
    __bf16* wqT = (__bf16*)alloc(D * D * 2);
    __bf16* wkT = (__bf16*)alloc(D * D * 2);
    __bf16* wpT = (__bf16*)alloc(D * D * 2);
    __bf16* wfT = (__bf16*)alloc(D * D * 2);
    float*  gq  = (float*) alloc(8 * D * 4);

    cvt_x_kernel<<<(int)(R * D / 8 / 256), 256, 0, stream>>>(x, xb, (int)(R * D / 8));
    cvt_wT_kernel<<<(int)(D * D / 256), 256, 0, stream>>>(wq, wqT);
    cvt_wT_kernel<<<(int)(D * D / 256), 256, 0, stream>>>(wk, wkT);
    cvt_wT_kernel<<<(int)(D * D / 256), 256, 0, stream>>>(wp, wpT);
    cvt_wT_kernel<<<(int)(D * D / 256), 256, 0, stream>>>(wf, wfT);
    zero_kernel<<<16, 256, 0, stream>>>(gq, 8 * 512);

    dim3 grid((unsigned)(R / 32)), block(256);
    gemm512_kernel<MODE_QNORM><<<grid, block, 0, stream>>>(
        xb, wqT, bq, nullptr, gq, nullptr, qn, nullptr);
    gemm512_kernel<MODE_KNORM><<<grid, block, 0, stream>>>(
        xb, wkT, bk, nullptr, nullptr, nullptr, nullptr, kn);
    gemm512_kernel<MODE_GEMM2><<<grid, block, 0, stream>>>(
        kn, wpT, bp, gq, nullptr, qn, nullptr, tmp);
    gemm512_kernel<MODE_GEMM3><<<grid, block, 0, stream>>>(
        tmp, wfT, bfv, nullptr, nullptr, nullptr, (float*)d_out, nullptr);
}
